// Model_38113539785431
// MI455X (gfx1250) — matrix operand order repaired and checked by reading, not compiled, not run on hardware
//
#include <hip/hip_runtime.h>

typedef __attribute__((ext_vector_type(16))) _Float16 v16h;
typedef __attribute__((ext_vector_type(8)))  _Float16 v8h;
typedef __attribute__((ext_vector_type(2)))  _Float16 h2;
typedef __attribute__((ext_vector_type(8)))  float    v8f;
typedef __attribute__((ext_vector_type(4)))  unsigned int u32x4;
typedef __attribute__((ext_vector_type(4)))  int      v4i;

union V16 { v16h v; v8h h[2]; };
union V8d { v8h v; h2 q[4]; };

constexpr int K       = 4096;
constexpr int N       = 11008;
constexpr int M       = 4 * 2048;     // BATCH * SEQ
constexpr int NGROUPS = 32;           // K / 128

constexpr int BM = 128, BN = 128, BK = 64;
constexpr int KSTEPS = K / BK;        // 64
constexpr int LDP = 72;               // padded LDS row (halves): 144B stride

// ---- CDNA5 async memory->LDS path selection ----
#if defined(__gfx1250__)
#define ON_DEVICE 1
#else
#define ON_DEVICE 0
#endif

#if ON_DEVICE && __has_builtin(__builtin_amdgcn_global_load_async_to_lds_b128)
#define ASYNC_MODE 1          // clang builtin
#elif ON_DEVICE
#define ASYNC_MODE 2          // inline asm fallback (documented CDNA5 op)
#else
#define ASYNC_MODE 0          // host-parse / generic fallback
#endif

#if ASYNC_MODE
#define WAIT_ASYNC(n) asm volatile("s_wait_asynccnt " #n ::: "memory")
#else
#define WAIT_ASYNC(n)
#endif

#if ASYNC_MODE == 1
typedef __attribute__((address_space(1))) v4i as1_v4i;  // global int4*
typedef __attribute__((address_space(3))) v4i as3_v4i;  // LDS int4*
#endif

__global__ __launch_bounds__(256)
void w4a16_wmma_gemm(const _Float16* __restrict__ x,
                     const int*      __restrict__ wq,
                     const _Float16* __restrict__ scales,
                     _Float16*       __restrict__ out) {
  __shared__ _Float16 lA[2][BM][LDP];  // activations, double-buffered, [m][k]
  __shared__ _Float16 lB[BN][LDP];     // dequantized weights, [n][k]

  const int tid   = threadIdx.x;
  const int mTile = blockIdx.x * BM;
  const int nTile = blockIdx.y * BN;

  // ---- loader mapping: 2 threads per row, 32 f16 (64B) each ----
  const int ldRow  = tid >> 1;       // 0..127
  const int ldHalf = tid & 1;        // 0..1

  const _Float16* gA0 = x      + (size_t)(mTile + ldRow) * K       + ldHalf * 32;
  const int*      gW0 = wq     + (size_t)(nTile + ldRow) * (K / 2) + ldHalf * 16;
  const _Float16* gS0 = scales + (size_t)(nTile + ldRow) * NGROUPS;

  u32x4 wReg[4];     // 16 packed bytes (one per int32)
  float sReg;        // group scale

  auto loadB = [&](int ks) {
    const u32x4* pw = (const u32x4*)(gW0 + (size_t)ks * (BK / 2));
#pragma unroll
    for (int i = 0; i < 4; ++i) wReg[i] = pw[i];
    sReg = (float)gS0[ks >> 1];      // group = (ks*64)/128
  };

  // A tile: direct memory -> LDS copy (async on CDNA5), 4 x b128 per thread
  auto copyA = [&](int ks, int buf) {
    const _Float16* src = gA0 + (size_t)ks * BK;
    _Float16*       dst = &lA[buf][ldRow][ldHalf * 32];
#if ASYNC_MODE == 1
    as1_v4i* g = (as1_v4i*)(unsigned long long)src;
    as3_v4i* l = (as3_v4i*)(unsigned int)(unsigned long long)dst;
    __builtin_amdgcn_global_load_async_to_lds_b128(g, l, 0,  0);
    __builtin_amdgcn_global_load_async_to_lds_b128(g, l, 16, 0);
    __builtin_amdgcn_global_load_async_to_lds_b128(g, l, 32, 0);
    __builtin_amdgcn_global_load_async_to_lds_b128(g, l, 48, 0);
#elif ASYNC_MODE == 2
    unsigned long long ga = (unsigned long long)src;
    unsigned int       la = (unsigned int)(unsigned long long)dst;
#pragma unroll
    for (int i = 0; i < 4; ++i)
      asm volatile("global_load_async_to_lds_b128 %0, %1, off"
                   :: "v"(la + i * 16), "v"(ga + i * 16) : "memory");
#else
#pragma unroll
    for (int i = 0; i < 4; ++i)
      *(v8h*)(dst + i * 8) = *(const v8h*)(src + i * 8);
#endif
  };

  // ---- compute mapping: 8 waves in a 4(M) x 2(N) grid ----
  const int lane = tid & 31;
  const int wave = tid >> 5;
  const int wm   = wave >> 1;        // 0..3 -> 32-row slab
  const int wn   = wave & 1;         // 0..1 -> 64-col slab
  const int lr   = lane & 15;        // row/col within fragment
  const int lh   = lane >> 4;        // half-select per ISA layout

  v8f acc[2][4] = {};

  copyA(0, 0);
  loadB(0);

  for (int ks = 0; ks < KSTEPS; ++ks) {
    __syncthreads();                 // all waves done reading lB / lA[(ks+1)&1]

    // stage B: unpack nibbles with f16 exponent trick, packed FMA dequant
    {
      _Float16* dstB = &lB[ldRow][ldHalf * 32];
      const _Float16 sh = (_Float16)sReg;
      const _Float16 bh = (_Float16)(-1032.0f * sReg);  // s*(1024+v) - 1032*s = s*(v-8)
      const h2 s2 = {sh, sh};
      const h2 b2 = {bh, bh};
#pragma unroll
      for (int i = 0; i < 4; ++i) {
        V8d d;
#pragma unroll
        for (int j = 0; j < 4; ++j) {
          const unsigned w = (unsigned)wReg[i][j];
          const unsigned p = 0x64006400u | (w & 0xFu) | ((w & 0xF0u) << 12);
          d.q[j] = __builtin_elementwise_fma(__builtin_bit_cast(h2, p), s2, b2);
        }
        *(v8h*)(dstB + i * 8) = d.v;
      }
    }

    if (ks + 1 < KSTEPS) {
      copyA(ks + 1, (ks + 1) & 1);   // async: overlaps this iteration's compute
      loadB(ks + 1);
      WAIT_ASYNC(0x4);               // in-order returns: tile ks is in LDS
    } else {
      WAIT_ASYNC(0x0);
    }
    __syncthreads();                 // lB stores + everyone's A-tile copy visible

    const int abuf = ks & 1;
#pragma unroll
    for (int kk = 0; kk < BK; kk += 32) {
      // A fragments: 16x32 f16, lanes 0-15: K 0..7 & 16..23; lanes 16-31: +8
      V16 af[2];
#pragma unroll
      for (int mi = 0; mi < 2; ++mi) {
        const _Float16* ar = &lA[abuf][wm * 32 + mi * 16 + lr][kk + lh * 8];
        af[mi].h[0] = *(const v8h*)ar;
        af[mi].h[1] = *(const v8h*)(ar + 16);
      }
      // B fragments: 32x16 f16, lane = col, K 0..7 & 16..23; lanes 16-31: +8
      V16 bf[4];
#pragma unroll
      for (int ni = 0; ni < 4; ++ni) {
        const _Float16* br = &lB[wn * 64 + ni * 16 + lr][kk + lh * 8];
        bf[ni].h[0] = *(const v8h*)br;
        bf[ni].h[1] = *(const v8h*)(br + 16);
      }
#pragma unroll
      for (int mi = 0; mi < 2; ++mi)
#pragma unroll
        for (int ni = 0; ni < 4; ++ni)
          acc[mi][ni] = __builtin_amdgcn_wmma_f32_16x16x32_f16(
              false, af[mi].v, false, bf[ni].v,
              (short)0, acc[mi][ni], false, false);
    }
  }

  // ---- epilogue: C/D layout -> global f16 ----
#pragma unroll
  for (int mi = 0; mi < 2; ++mi) {
    const int m0 = mTile + wm * 32 + mi * 16 + lh * 8;   // VGPR r holds M = r (+8 hi lanes)
#pragma unroll
    for (int ni = 0; ni < 4; ++ni) {
      const int n0 = nTile + wn * 64 + ni * 16 + lr;
      _Float16* op = out + (size_t)m0 * N + n0;
#pragma unroll
      for (int r = 0; r < 8; ++r)
        op[(size_t)r * N] = (_Float16)acc[mi][ni][r];
    }
  }
}

extern "C" void kernel_launch(void* const* d_in, const int* in_sizes, int n_in,
                              void* d_out, int out_size, void* d_ws, size_t ws_size,
                              hipStream_t stream) {
  (void)in_sizes; (void)n_in; (void)out_size; (void)d_ws; (void)ws_size;
  const _Float16* x  = (const _Float16*)d_in[0];
  const int*      wq = (const int*)d_in[1];
  const _Float16* sc = (const _Float16*)d_in[2];
  _Float16*       o  = (_Float16*)d_out;

  dim3 grid(M / BM, N / BN);   // 64 x 86
  dim3 block(256);             // 8 wave32
  w4a16_wmma_gemm<<<grid, block, 0, stream>>>(x, wq, sc, o);
}
